// DropoutMM_33887291966046
// MI455X (gfx1250) — matrix operand order repaired and checked by reading, not compiled, not run on hardware
//
#include <hip/hip_runtime.h>

typedef __attribute__((ext_vector_type(16))) __bf16 v16bf;
typedef __attribute__((ext_vector_type(8)))  __bf16 v8bf;
typedef __attribute__((ext_vector_type(4)))  __bf16 v4bf;
typedef __attribute__((ext_vector_type(8)))  float  v8f;

static constexpr int M_DIM = 8192;
static constexpr int K_DIM = 4096;
static constexpr int N_DIM = 4096;
static constexpr int BLK   = 128;                 // dropout block size
static constexpr int KB    = K_DIM / BLK;         // 32 mask columns
static constexpr int LDA   = 40;                  // LDS row stride (32 + 8 pad) in bf16
static constexpr float SCALE = 1.0f / 0.9f;       // 1/(1-p)

__device__ __forceinline__ v16bf cat8(v8bf a, v8bf b) {
  return __builtin_shufflevector(a, b, 0,1,2,3,4,5,6,7,8,9,10,11,12,13,14,15);
}

__global__ __launch_bounds__(256, 2)
void dropout_gemm_wmma(const float* __restrict__ x,
                       const float* __restrict__ w,
                       const int*   __restrict__ mask,
                       float*       __restrict__ out)
{
  // bf16 hi/lo staging buffers: 4 * 128*40*2B = 40 KB LDS
  __shared__ alignas(16) __bf16 sAh[BLK * LDA];
  __shared__ alignas(16) __bf16 sAl[BLK * LDA];
  __shared__ alignas(16) __bf16 sBh[BLK * LDA];
  __shared__ alignas(16) __bf16 sBl[BLK * LDA];

  const int tid     = threadIdx.x;
  const int lane    = tid & 31;
  const int wave    = tid >> 5;
  const int mWave   = (wave >> 1) * 32;   // 4 waves along M: 32 rows each
  const int nWave   = (wave & 1) * 64;    // 2 waves along N: 64 cols each
  const int lrow    = lane & 15;
  const int halfSel = lane >> 4;          // 0: lanes 0-15, 1: lanes 16-31

  const int mBase = blockIdx.y * BLK;
  const int nBase = blockIdx.x * BLK;

  v8f acc[2][4] = {};                     // 2 M-tiles x 4 N-tiles of 16x16 f32

  for (int kb = 0; kb < KB; ++kb) {
    const int keep = mask[blockIdx.y * KB + kb];   // uniform scalar load
    if (keep == 0) continue;                        // skip dropped 128-K block

    #pragma unroll 1
    for (int kk = 0; kk < 4; ++kk) {
      const int k0 = kb * BLK + kk * 32;

      __syncthreads();   // prior iteration's LDS reads complete

      // ---- cooperative load + fp32 -> (bf16 hi, bf16 lo) split into LDS ----
      #pragma unroll
      for (int i = 0; i < 4; ++i) {
        const int idx = tid + i * 256;       // 1024 float4 per 128x32 panel
        const int row = idx >> 3;
        const int c4  = (idx & 7) * 4;       // element column (multiple of 4)

        // A panel (x), scaled by 1/(1-p)
        {
          const float4 v = *(const float4*)(x + (size_t)(mBase + row) * K_DIM + k0 + c4);
          const float f[4] = {v.x * SCALE, v.y * SCALE, v.z * SCALE, v.w * SCALE};
          v4bf h, l;
          #pragma unroll
          for (int j = 0; j < 4; ++j) {
            h[j] = (__bf16)f[j];
            l[j] = (__bf16)(f[j] - (float)h[j]);
          }
          *(v4bf*)&sAh[row * LDA + c4] = h;
          *(v4bf*)&sAl[row * LDA + c4] = l;
        }
        // B panel (weight rows = N, contiguous in K)
        {
          const float4 v = *(const float4*)(w + (size_t)(nBase + row) * K_DIM + k0 + c4);
          const float f[4] = {v.x, v.y, v.z, v.w};
          v4bf h, l;
          #pragma unroll
          for (int j = 0; j < 4; ++j) {
            h[j] = (__bf16)f[j];
            l[j] = (__bf16)(f[j] - (float)h[j]);
          }
          *(v4bf*)&sBh[row * LDA + c4] = h;
          *(v4bf*)&sBl[row * LDA + c4] = l;
        }
      }

      // prefetch next K chunk while we compute this one
      {
        int kn = k0 + 32;
        if (kn >= K_DIM) kn = k0;
        const int prow = tid >> 1;
        __builtin_prefetch(x + (size_t)(mBase + prow) * K_DIM + kn, 0, 1);
        __builtin_prefetch(w + (size_t)(nBase + prow) * K_DIM + kn, 0, 1);
      }

      __syncthreads();   // LDS panels ready

      // ---- A fragments (ISA 16-bit A 16x32 layout):
      //  lanes 0-15: M=lrow, K={0..7,16..23}; lanes 16-31: M=lrow, K={8..15,24..31}
      v16bf Ah[2], Al[2];
      #pragma unroll
      for (int mt = 0; mt < 2; ++mt) {
        const int r  = mWave + mt * 16 + lrow;
        const int kc = halfSel * 8;
        Ah[mt] = cat8(*(const v8bf*)&sAh[r * LDA + kc],
                      *(const v8bf*)&sAh[r * LDA + kc + 16]);
        Al[mt] = cat8(*(const v8bf*)&sAl[r * LDA + kc],
                      *(const v8bf*)&sAl[r * LDA + kc + 16]);
      }

      // ---- B fragments (B 32x16): lanes 0-15: N=lrow, K={0..7,16..23}; lanes 16-31: K={8..15,24..31}
      v16bf Bh[4], Bl[4];
      #pragma unroll
      for (int nt = 0; nt < 4; ++nt) {
        const int c  = nWave + nt * 16 + lrow;
        const int kc = halfSel * 8;
        Bh[nt] = cat8(*(const v8bf*)&sBh[c * LDA + kc],
                      *(const v8bf*)&sBh[c * LDA + kc + 16]);
        Bl[nt] = cat8(*(const v8bf*)&sBl[c * LDA + kc],
                      *(const v8bf*)&sBl[c * LDA + kc + 16]);
      }

      // ---- 3-term bf16 split-GEMM: Ah*Bh + Ah*Bl + Al*Bh ----
      // consecutive WMMAs hit different accumulators (independent chains)
      #pragma unroll
      for (int mt = 0; mt < 2; ++mt)
        #pragma unroll
        for (int nt = 0; nt < 4; ++nt)
          acc[mt][nt] = __builtin_amdgcn_wmma_f32_16x16x32_bf16(
              false, Ah[mt], false, Bh[nt], (short)0, acc[mt][nt], false, false);
      #pragma unroll
      for (int mt = 0; mt < 2; ++mt)
        #pragma unroll
        for (int nt = 0; nt < 4; ++nt)
          acc[mt][nt] = __builtin_amdgcn_wmma_f32_16x16x32_bf16(
              false, Ah[mt], false, Bl[nt], (short)0, acc[mt][nt], false, false);
      #pragma unroll
      for (int mt = 0; mt < 2; ++mt)
        #pragma unroll
        for (int nt = 0; nt < 4; ++nt)
          acc[mt][nt] = __builtin_amdgcn_wmma_f32_16x16x32_bf16(
              false, Al[mt], false, Bh[nt], (short)0, acc[mt][nt], false, false);
    }
  }

  // ---- store: D 16x16 f32 layout: M = vgpr + 8*halfSel, N = lrow ----
  #pragma unroll
  for (int mt = 0; mt < 2; ++mt) {
    #pragma unroll
    for (int nt = 0; nt < 4; ++nt) {
      float* p = out + (size_t)(mBase + mWave + mt * 16 + 8 * halfSel) * N_DIM
                     + nBase + nWave + nt * 16 + lrow;
      #pragma unroll
      for (int r = 0; r < 8; ++r)
        p[(size_t)r * N_DIM] = acc[mt][nt][r];
    }
  }
}

extern "C" void kernel_launch(void* const* d_in, const int* in_sizes, int n_in,
                              void* d_out, int out_size, void* d_ws, size_t ws_size,
                              hipStream_t stream) {
  const float* x    = (const float*)d_in[0];
  const float* w    = (const float*)d_in[1];
  const int*   mask = (const int*)d_in[2];
  float*       out  = (float*)d_out;

  dim3 grid(N_DIM / BLK, M_DIM / BLK);   // (32, 64)
  dropout_gemm_wmma<<<grid, dim3(256), 0, stream>>>(x, w, mask, out);
}
